// LUTLinear_t_69827578298580
// MI455X (gfx1250) — matrix operand order repaired and checked by reading, not compiled, not run on hardware
//
#include <hip/hip_runtime.h>
#include <hip/hip_bf16.h>
#include <stdint.h>

typedef __attribute__((ext_vector_type(16))) _Float16     v16h;
typedef __attribute__((ext_vector_type(8)))  float        v8f;
typedef __attribute__((ext_vector_type(4)))  int          v4i;
typedef __attribute__((ext_vector_type(8)))  unsigned int v8u;

#define CB    128   // number of codebooks (IN / VLEN)
#define KCENT 16    // centroids per codebook
#define VLEN  16    // sub-vector length
#define NPAIR (CB / 2)

#if defined(__has_builtin)
#  if __has_builtin(__builtin_amdgcn_global_load_async_to_lds_b128) && \
      __has_builtin(__builtin_amdgcn_s_wait_asynccnt)
#    define USE_ASYNC_LDS 1
#  endif
#endif
#ifndef USE_ASYNC_LDS
#  define USE_ASYNC_LDS 0
#endif

// ---------------------------------------------------------------------------
// Kernel 1: build LUT in f16, B-fragment-friendly layout lutT[c][o][k]
//   lut[c][k][o] = sum_v cents[c][k][v] * W[c*VLEN+v][o]   (f32 accumulate)
// ---------------------------------------------------------------------------
__global__ void __launch_bounds__(256)
build_lut_kernel(const float* __restrict__ W,
                 const float* __restrict__ cents,
                 uint4* __restrict__ lutQ,
                 int OUT)
{
    __shared__ float sc[KCENT * VLEN];
    const int c = blockIdx.x;
    const int o = blockIdx.y * 256 + threadIdx.x;
    sc[threadIdx.x] = cents[c * KCENT * VLEN + threadIdx.x];
    __syncthreads();

    float acc[KCENT];
#pragma unroll
    for (int k = 0; k < KCENT; ++k) acc[k] = 0.0f;

#pragma unroll
    for (int v = 0; v < VLEN; ++v) {
        float wv = W[(size_t)(c * VLEN + v) * OUT + o];
#pragma unroll
        for (int k = 0; k < KCENT; ++k)
            acc[k] = fmaf(sc[k * VLEN + v], wv, acc[k]);
    }

    union { _Float16 h[16]; uint4 q[2]; } u;
#pragma unroll
    for (int k = 0; k < KCENT; ++k) u.h[k] = (_Float16)acc[k];

    size_t base = ((size_t)c * OUT + o) * 2;
    lutQ[base + 0] = u.q[0];
    lutQ[base + 1] = u.q[1];
}

// ---------------------------------------------------------------------------
// Kernel 2: Chebyshev (L-inf) nearest-centroid indices, first-min tie break.
// ---------------------------------------------------------------------------
__global__ void __launch_bounds__(256)
argmin_kernel(const float* __restrict__ x,
              const float* __restrict__ cents,
              uint8_t* __restrict__ idx8,
              int IN)
{
    __shared__ float sc[KCENT * VLEN];
    const int c = blockIdx.x;
    const int n = blockIdx.y * 256 + threadIdx.x;
    sc[threadIdx.x] = cents[c * KCENT * VLEN + threadIdx.x];
    __syncthreads();

    float xv[VLEN];
#pragma unroll
    for (int v = 0; v < VLEN; ++v)
        xv[v] = x[(size_t)n * IN + c * VLEN + v];

    float best = 3.0e38f;
    int   bi   = 0;
#pragma unroll
    for (int k = 0; k < KCENT; ++k) {
        float d = 0.0f;
#pragma unroll
        for (int v = 0; v < VLEN; ++v)
            d = fmaxf(d, fabsf(xv[v] - sc[k * VLEN + v]));
        if (d < best) { best = d; bi = k; }   // strict '<' keeps first minimum
    }
    idx8[(size_t)n * CB + c] = (uint8_t)bi;
}

// ---------------------------------------------------------------------------
// One-hot f16 row segment (8 halves = 4 dwords) via 64-bit shifts.
// ---------------------------------------------------------------------------
__device__ __forceinline__ void onehot_half8(int v, int kbase, uint32_t* w)
{
    unsigned h  = (unsigned)(v - kbase);
    unsigned h4 = h - 4u;
    uint64_t lo = (h  < 4u) ? (0x3C00ull << (h  * 16u)) : 0ull;
    uint64_t hi = (h4 < 4u) ? (0x3C00ull << (h4 * 16u)) : 0ull;
    w[0] = (uint32_t)lo; w[1] = (uint32_t)(lo >> 32);
    w[2] = (uint32_t)hi; w[3] = (uint32_t)(hi >> 32);
}

// ---------------------------------------------------------------------------
// Kernel 3: one-hot gather GEMM via v_wmma_f32_16x16x32_f16 + bias.
// Block: 256 threads (8 waves); tile 128 tokens x 128 out columns.
// __launch_bounds__(256, 2) raises the VGPR budget so the ~160-reg working
// set (64 acc + 64 live B fragments + A) fits without scratch spills.
// ---------------------------------------------------------------------------
__global__ void __launch_bounds__(256, 2)
gather_wmma_kernel(const uint4* __restrict__ lutQ,
                   const unsigned short* __restrict__ idx16,
                   const float* __restrict__ bias,
                   float* __restrict__ out,
                   int OUT)
{
    __shared__ uint4 ldsB[2][512];   // 2 x 8KB: [buf][(cb*128 + o) * 2 + part]

    const int tid   = threadIdx.x;
    const int lane  = tid & 31;
    const int wave  = tid >> 5;
    const int ob    = blockIdx.x * 128;
    const int tb    = blockIdx.y * 128;
    const int hi    = lane >> 4;          // 0: lanes 0-15, 1: lanes 16-31
    const int kbase = hi * 8;
    const int t     = tb + wave * 16 + (lane & 15);

    // staging: thread stages 32B for unit (cb = tid>>7, o_local = tid&127)
    const size_t qoff = (size_t)(tid >> 7) * (size_t)OUT * 2
                      + ((size_t)ob + (size_t)(tid & 127)) * 2;

    // ---- prologue: stage pair 0 into buffer 0, preload idx pair 0 ----
    unsigned pair = idx16[(size_t)t * NPAIR];
    {
        const uint4* src = lutQ + qoff;
#if USE_ASYNC_LDS
        __builtin_amdgcn_global_load_async_to_lds_b128(
            (v4i*)(src), (v4i*)&ldsB[0][tid * 2], 0, 0);
        __builtin_amdgcn_global_load_async_to_lds_b128(
            (v4i*)(src + 1), (v4i*)&ldsB[0][tid * 2 + 1], 0, 0);
        __builtin_amdgcn_s_wait_asynccnt(0);
#else
        uint4 r0 = src[0], r1 = src[1];
        ldsB[0][tid * 2]     = r0;
        ldsB[0][tid * 2 + 1] = r1;
#endif
        __syncthreads();
    }

    v8f acc[8] = {};

#pragma unroll 2
    for (int c = 0; c < NPAIR; ++c) {
        const int cur  = c & 1;
        const int next = (c + 1 < NPAIR) ? c + 1 : c;   // clamped: branch-free

        // ---- kick off staging of pair `next` into the other buffer ----
#if USE_ASYNC_LDS
        {
            const uint4* src = lutQ + (size_t)next * (size_t)OUT * 4 + qoff;
            __builtin_amdgcn_global_load_async_to_lds_b128(
                (v4i*)(src), (v4i*)&ldsB[cur ^ 1][tid * 2], 0, 0);
            __builtin_amdgcn_global_load_async_to_lds_b128(
                (v4i*)(src + 1), (v4i*)&ldsB[cur ^ 1][tid * 2 + 1], 0, 0);
        }
#else
        uint4 r0, r1;
        {
            const uint4* src = lutQ + (size_t)next * (size_t)OUT * 4 + qoff;
            r0 = src[0];
            r1 = src[1];
        }
#endif

        // ---- one-hot A fragment from the carried idx pair ----
        union { v16h h; uint32_t w[8]; } a;
        const int vsel = hi ? (int)(pair >> 8) : (int)(pair & 0xffu);
        onehot_half8(vsel, 0, a.w);
        onehot_half8(vsel, 8, a.w + 4);

        // ---- prefetch next idx pair (latency covered by B loads + WMMAs) ----
        pair = idx16[(size_t)t * NPAIR + next];

        // ---- load ALL 8 B fragments ----
        union Bf { v16h h; v8u u; uint4 q[2]; } b[8];
#pragma unroll
        for (int j = 0; j < 8; ++j) {
            const int row = hi * 128 + j * 16 + (lane & 15);
            b[j].q[0] = ldsB[cur][row * 2];
            b[j].q[1] = ldsB[cur][row * 2 + 1];
        }
        // scheduling fence: loads may not sink past this point; all 8
        // fragments stay live in distinct 8-VGPR groups.
        asm volatile(""
            : "+v"(b[0].u), "+v"(b[1].u), "+v"(b[2].u), "+v"(b[3].u),
              "+v"(b[4].u), "+v"(b[5].u), "+v"(b[6].u), "+v"(b[7].u));

        // ---- 8 back-to-back WMMAs ----
#pragma unroll
        for (int j = 0; j < 8; ++j)
            acc[j] = __builtin_amdgcn_wmma_f32_16x16x32_f16(
                         false, a.h, false, b[j].h, (short)0, acc[j], false, false);

        // ---- finish staging and publish the next buffer ----
#if USE_ASYNC_LDS
        __builtin_amdgcn_s_wait_asynccnt(0);
#else
        ldsB[cur ^ 1][tid * 2]     = r0;
        ldsB[cur ^ 1][tid * 2 + 1] = r1;
#endif
        __syncthreads();
    }

    // D layout: VGPR r holds M=r (lanes 0-15) / M=r+8 (lanes 16-31), N=lane%16
    const int col0 = ob + (lane & 15);
#pragma unroll
    for (int j = 0; j < 8; ++j) {
        const int col = col0 + j * 16;
        const float bj = bias[col];
#pragma unroll
        for (int r = 0; r < 8; ++r) {
            const int m = r + hi * 8;
            out[(size_t)(tb + wave * 16 + m) * OUT + col] = acc[j][r] + bj;
        }
    }
}

// ---------------------------------------------------------------------------
extern "C" void kernel_launch(void* const* d_in, const int* in_sizes, int n_in,
                              void* d_out, int out_size, void* d_ws, size_t ws_size,
                              hipStream_t stream)
{
    const float* x     = (const float*)d_in[0];
    const float* W     = (const float*)d_in[1];
    const float* cents = (const float*)d_in[2];
    const float* bias  = (const float*)d_in[3];

    const int OUT = in_sizes[3];            // 4096
    const int IN  = in_sizes[1] / OUT;      // 2048
    const int N   = in_sizes[0] / IN;       // 4096   (B*S)

    char* ws = (char*)d_ws;
    uint4* lutQ = (uint4*)ws;                                   // CB*OUT*16 f16 = 16 MB
    const size_t lutBytes = (size_t)CB * (size_t)OUT * KCENT * 2;
    uint8_t*        idx8  = (uint8_t*)(ws + lutBytes);          // N*CB bytes
    unsigned short* idx16 = (unsigned short*)idx8;

    float* out = (float*)d_out;

    build_lut_kernel<<<dim3(CB, OUT / 256), 256, 0, stream>>>(W, cents, lutQ, OUT);
    argmin_kernel  <<<dim3(CB, N / 256),   256, 0, stream>>>(x, cents, idx8, IN);
    gather_wmma_kernel<<<dim3(OUT / 128, N / 128), 256, 0, stream>>>(
        lutQ, idx16, bias, out, OUT);
}
